// KGCompletionGNN_47579647705219
// MI455X (gfx1250) — compile-verified, matrix operand order checked by reading, not run on hardware
//
#include <hip/hip_runtime.h>

typedef __bf16 bf16;
typedef __attribute__((ext_vector_type(16))) __bf16   v16bf;
typedef __attribute__((ext_vector_type(8)))  float    v8f;
typedef __attribute__((ext_vector_type(4)))  unsigned v4u;

#define D      128
#define TILE_M 16
#define LRELU(x) ((x) >= 0.f ? (x) : 0.01f * (x))

// koff(v,half): K offset of dword v within a 16-bit WMMA fragment (ISA 7.12.2)
__host__ __device__ __forceinline__ int frag_koff(int v, int half) {
  return ((v < 4) ? 2 * v : 2 * v + 8) + 8 * half;
}

// ---------------------------------------------------------------------------
// Weight packing, fragment-major: for k-step s, column c, lane-half h, the 8
// dwords {bf16 W[32s+koff(v,h)][c], bf16 W[32s+koff(v,h)+1][c]} are stored
// contiguously at dst[(((s*128)+c)*2+h)*8 + v]. A lane then fetches one whole
// B fragment with two global_load_b128 from a lane-constant base + s*8KB.
// Contiguous L layers handled by letting s run across layers (K % 32 == 0).
// ---------------------------------------------------------------------------
__global__ void pack_weights_frag_kernel(const float* __restrict__ src,
                                         unsigned* __restrict__ dst, int n) {
  int i = blockIdx.x * blockDim.x + threadIdx.x;
  if (i >= n) return;
  int v    = i & 7;
  int half = (i >> 3) & 1;
  int col  = (i >> 4) & 127;
  int s    = i >> 11;
  long k   = 32L * s + frag_koff(v, half);
  union { bf16 h[2]; unsigned u; } p;
  p.h[0] = (bf16)src[k * D + col];
  p.h[1] = (bf16)src[(k + 1) * D + col];
  dst[i] = p.u;
}

__global__ void fill_zero_kernel(float* __restrict__ p, long n) {
  long i = (long)blockIdx.x * blockDim.x + threadIdx.x;
  if (i < n) p[i] = 0.f;
}

// counts[n] = #occurrences of n in ht (both columns) — layer invariant.
__global__ void count_kernel(const int* __restrict__ ht,
                             float* __restrict__ cnt, int M) {
  int e = blockIdx.x * blockDim.x + threadIdx.x;
  if (e >= M) return;
  atomicAdd(&cnt[ht[2 * e]], 1.f);
  atomicAdd(&cnt[ht[2 * e + 1]], 1.f);
}

// ---------------------------------------------------------------------------
// Edge kernel: one block = 16 edges, 8 waves; wave w owns output cols
// [16w,16w+16).
//  1) stage [head|E|tail] (16x384) into LDS as bf16; mirror f32 E into Cbuf
//  2) eu GEMM (K=384), B fragments = 2x b128 from L2-resident packed weights
//  3) bias + leakyrelu + residual (Cbuf in place); cross-wave LayerNorm;
//     write E_out (f32) and overwrite LDS E slot with bf16 E'
//  4) fwd/back message GEMMs (K=256) and f32 atomic scatter-add into agg
// ---------------------------------------------------------------------------
__launch_bounds__(256)
__global__ void edge_kernel(const float* __restrict__ H,
                            const float* __restrict__ E_in,
                            float* __restrict__ E_out,
                            const int* __restrict__ ht,
                            const unsigned* __restrict__ Wpk_eu,
                            const float* __restrict__ bias_eu,
                            const float* __restrict__ g_e,
                            const float* __restrict__ beta_e,
                            const unsigned* __restrict__ Wpk_f,
                            const float* __restrict__ bias_f,
                            const unsigned* __restrict__ Wpk_b,
                            const float* __restrict__ bias_b,
                            float* __restrict__ agg, int M) {
  __shared__ __attribute__((aligned(16))) bf16 Abuf[TILE_M][3 * D];
  __shared__ float Cbuf[TILE_M][D + 4];
  __shared__ int   nidx[2][TILE_M];

  const int tid  = threadIdx.x;
  const int lane = tid & 31;
  const int w    = tid >> 5;
  const int e0   = blockIdx.x * TILE_M;

  if (tid < TILE_M) {
    int e = min(e0 + tid, M - 1);
    nidx[0][tid] = ht[2 * e];
    nidx[1][tid] = ht[2 * e + 1];
  }
  __syncthreads();

  // ---- Phase 1: stage triple tile (bf16); mirror E (f32) into Cbuf ------
  for (int i = tid; i < TILE_M * 3 * D; i += 256) {
    int row = i / (3 * D), col = i - row * 3 * D;
    long e = min(e0 + row, M - 1);
    float v;
    if (col < D) {
      v = H[(long)nidx[0][row] * D + col];
    } else if (col < 2 * D) {
      v = E_in[e * D + (col - D)];
      Cbuf[row][col - D] = v;            // keep f32 E for the residual
    } else {
      v = H[(long)nidx[1][row] * D + (col - 2 * D)];
    }
    Abuf[row][col] = (bf16)v;
  }
  __syncthreads();

  const int m    = lane & 15;      // A-row within tile / B,C column lane
  const int half = lane >> 4;
  const int ncol = w * 16 + m;     // absolute output column (0..127)
  const int mrow = half * 8;       // C-fragment base row

  union AF { v16bf v; unsigned u[8]; v4u q[2]; };

  // Lane-constant fragment bases: dword ((ncol*2+half)*8) -> v4u index *2
  const size_t laneB = (size_t)((ncol << 1) + half) * 2;
  const v4u* BpEU = (const v4u*)Wpk_eu + laneB;
  const v4u* BpF  = (const v4u*)Wpk_f  + laneB;
  const v4u* BpB  = (const v4u*)Wpk_b  + laneB;

  // ---- Phase 2: eu = triple @ W_eu --------------------------------------
  v8f acc = {};
#pragma unroll
  for (int s = 0; s < 12; ++s) {
    AF a, b;
#pragma unroll
    for (int v = 0; v < 8; ++v)
      a.u[v] = *(const unsigned*)&Abuf[m][32 * s + frag_koff(v, half)];
    b.q[0] = BpEU[(size_t)s * 512];
    b.q[1] = BpEU[(size_t)s * 512 + 1];
    acc = __builtin_amdgcn_wmma_f32_16x16x32_bf16(
        false, a.v, false, b.v, (short)0, acc, false, false);
  }

  // ---- Phase 3: bias + lrelu + residual, then LayerNorm -----------------
  {
    float be = bias_eu[ncol];
#pragma unroll
    for (int r = 0; r < 8; ++r) {
      int row = mrow + r;
      float v = acc[r] + be;
      v = LRELU(v);
      v += Cbuf[row][ncol];              // f32 E residual from LDS
      Cbuf[row][ncol] = v;
    }
  }
  __syncthreads();

  {
    int row = tid >> 4;
    int sub = tid & 15;
    float s = 0.f, ss = 0.f, vals[8];
#pragma unroll
    for (int j = 0; j < 8; ++j) {
      float v = Cbuf[row][sub + 16 * j];
      vals[j] = v; s += v; ss += v * v;
    }
#pragma unroll
    for (int off = 8; off >= 1; off >>= 1) {
      s  += __shfl_xor(s, off, 16);
      ss += __shfl_xor(ss, off, 16);
    }
    float mu   = s * (1.f / D);
    float var  = ss * (1.f / D) - mu * mu;
    float rsig = rsqrtf(var + 1e-5f);
    long e = e0 + row;
    bool valid = e < M;
#pragma unroll
    for (int j = 0; j < 8; ++j) {
      int n = sub + 16 * j;
      float o = (vals[j] - mu) * rsig * g_e[n] + beta_e[n];
      if (valid) E_out[e * D + n] = o;
      Abuf[row][D + n] = (bf16)o;        // E slot <- E' (bf16) for msg GEMMs
    }
  }
  __syncthreads();

  // ---- Phase 4: message GEMMs + scatter ---------------------------------
  v8f accf = {}, accb = {};
#pragma unroll
  for (int s = 0; s < 8; ++s) {
    AF af, ab, bfm, bbm;
#pragma unroll
    for (int v = 0; v < 8; ++v) {
      int kk = 32 * s + frag_koff(v, half);
      af.u[v] = *(const unsigned*)&Abuf[m][kk];                 // head | E'
      int ak  = (kk < D) ? (kk + 2 * D) : kk;                   // tail | E'
      ab.u[v] = *(const unsigned*)&Abuf[m][ak];
    }
    bfm.q[0] = BpF[(size_t)s * 512];
    bfm.q[1] = BpF[(size_t)s * 512 + 1];
    bbm.q[0] = BpB[(size_t)s * 512];
    bbm.q[1] = BpB[(size_t)s * 512 + 1];
    accf = __builtin_amdgcn_wmma_f32_16x16x32_bf16(
        false, af.v, false, bfm.v, (short)0, accf, false, false);
    accb = __builtin_amdgcn_wmma_f32_16x16x32_bf16(
        false, ab.v, false, bbm.v, (short)0, accb, false, false);
  }

  float bfb = bias_f[ncol];
  float bbb = bias_b[ncol];
#pragma unroll
  for (int r = 0; r < 8; ++r) {
    int row = mrow + r;
    if (e0 + row < M) {
      int dstf = nidx[1][row];   // fwd msgs -> tail node
      int dstb = nidx[0][row];   // back msgs -> head node
      atomicAdd(&agg[(long)dstf * D + ncol], accf[r] + bfb);
      atomicAdd(&agg[(long)dstb * D + ncol], accb[r] + bbb);
    }
  }
}

// ---------------------------------------------------------------------------
// Node update: H = LN(leakyrelu(agg/max(cnt,1)) + H). One wave per node row.
// Safe in-place (row-wise).
// ---------------------------------------------------------------------------
__launch_bounds__(256)
__global__ void node_kernel(const float* __restrict__ Hin,
                            const float* __restrict__ agg,
                            const float* __restrict__ cnt,
                            const float* __restrict__ g,
                            const float* __restrict__ beta,
                            float* __restrict__ Hout, int N) {
  int row  = blockIdx.x * 8 + (threadIdx.x >> 5);
  int lane = threadIdx.x & 31;
  if (row >= N) return;
  float inv = 1.f / fmaxf(cnt[row], 1.f);
  float v[4], s = 0.f, ss = 0.f;
#pragma unroll
  for (int j = 0; j < 4; ++j) {
    int n = lane + 32 * j;
    float a = agg[(long)row * D + n] * inv;
    a = LRELU(a);
    a += Hin[(long)row * D + n];
    v[j] = a; s += a; ss += a * a;
  }
#pragma unroll
  for (int off = 16; off >= 1; off >>= 1) {
    s  += __shfl_xor(s, off, 32);
    ss += __shfl_xor(ss, off, 32);
  }
  float mu   = s * (1.f / D);
  float var  = ss * (1.f / D) - mu * mu;
  float rsig = rsqrtf(var + 1e-5f);
#pragma unroll
  for (int j = 0; j < 4; ++j) {
    int n = lane + 32 * j;
    Hout[(long)row * D + n] = (v[j] - mu) * rsig * g[n] + beta[n];
  }
}

// ---------------------------------------------------------------------------
extern "C" void kernel_launch(void* const* d_in, const int* in_sizes, int n_in,
                              void* d_out, int out_size, void* d_ws,
                              size_t ws_size, hipStream_t stream) {
  const float* H    = (const float*)d_in[0];
  const float* E    = (const float*)d_in[1];
  const int*   ht   = (const int*)d_in[2];
  // d_in[3] = queries (unused by reference output)
  const float* W_eu = (const float*)d_in[4];
  const float* b_eu = (const float*)d_in[5];
  const float* g_e  = (const float*)d_in[6];
  const float* be_e = (const float*)d_in[7];
  const float* W_f  = (const float*)d_in[8];
  const float* b_f  = (const float*)d_in[9];
  const float* W_b  = (const float*)d_in[10];
  const float* b_b  = (const float*)d_in[11];
  const float* g_h  = (const float*)d_in[12];
  const float* be_h = (const float*)d_in[13];

  const int N  = in_sizes[0] / D;            // 100000
  const int M  = in_sizes[1] / D;            // 500000
  const int Lw = in_sizes[4] / (3 * D * D);  // 2

  // Workspace carve-out (256B aligned)
  char* ws = (char*)d_ws;
  size_t off = 0;
  auto carve = [&](size_t bytes) -> void* {
    void* p = ws + off;
    off = (off + bytes + 255) & ~(size_t)255;
    return p;
  };
  unsigned* Wpk_eu = (unsigned*)carve((size_t)Lw * 192 * D * 4);
  unsigned* Wpk_f  = (unsigned*)carve((size_t)Lw * 128 * D * 4);
  unsigned* Wpk_b  = (unsigned*)carve((size_t)Lw * 128 * D * 4);
  float*    cnt    = (float*)carve((size_t)N * 4);
  float*    agg    = (float*)carve((size_t)N * D * 4);
  float*    E_w    = (float*)carve((size_t)M * D * 4);

  // Pack weights to fragment-major bf16 dwords (once per launch)
  {
    int n1 = Lw * 12 * 2048;   // eu: K=384 -> 12 k-steps/layer
    pack_weights_frag_kernel<<<(n1 + 255) / 256, 256, 0, stream>>>(W_eu,
                                                                   Wpk_eu, n1);
    int n2 = Lw * 8 * 2048;    // fwd/back: K=256 -> 8 k-steps/layer
    pack_weights_frag_kernel<<<(n2 + 255) / 256, 256, 0, stream>>>(W_f,
                                                                   Wpk_f, n2);
    pack_weights_frag_kernel<<<(n2 + 255) / 256, 256, 0, stream>>>(W_b,
                                                                   Wpk_b, n2);
  }

  // Degree counts (layer invariant)
  fill_zero_kernel<<<(N + 255) / 256, 256, 0, stream>>>(cnt, N);
  count_kernel<<<(M + 255) / 256, 256, 0, stream>>>(ht, cnt, M);

  long aggN = (long)N * D;
  for (int l = 0; l < Lw; ++l) {
    fill_zero_kernel<<<(int)((aggN + 255) / 256), 256, 0, stream>>>(agg, aggN);
    const float* Hin = (l == 0) ? H : (const float*)d_out;
    const float* Ein = (l == 0) ? E : (const float*)E_w;
    edge_kernel<<<(M + TILE_M - 1) / TILE_M, 256, 0, stream>>>(
        Hin, Ein, E_w, ht,
        Wpk_eu + (size_t)l * 192 * D, b_eu + l * D, g_e + l * D, be_e + l * D,
        Wpk_f + (size_t)l * 128 * D, b_f + l * D,
        Wpk_b + (size_t)l * 128 * D, b_b + l * D,
        agg, M);
    node_kernel<<<(N + 7) / 8, 256, 0, stream>>>(
        Hin, agg, cnt, g_h + l * D, be_h + l * D, (float*)d_out, N);
  }
}
